// MLAAttention_74071005987524
// MI455X (gfx1250) — matrix operand order repaired and checked by reading, not compiled, not run on hardware
//
#include <hip/hip_runtime.h>
#include <hip/hip_bf16.h>

// ---------------------------------------------------------------------------
// MLA forward for MI455X (gfx1250, wave32, WMMA 16x16x32 bf16).
// GEMMs: 4-wave 64x32 tiles, B panel staged in LDS via
// global_load_async_to_lds_b128 (ASYNCcnt), double-buffered.
// ---------------------------------------------------------------------------

typedef __bf16 bf16_t;
typedef __bf16 v16bf __attribute__((ext_vector_type(16)));
typedef __bf16 v8bf  __attribute__((ext_vector_type(8)));
typedef float  v8f   __attribute__((ext_vector_type(8)));

#define DMODEL 2048
#define NQH    16
#define NKVH   4
#define DHC    128   // content head dim
#define DRR    64    // rope head dim
#define DKVL   512   // kv latent
#define DQCL   1536  // q latent
#define BBB    2
#define SEQ    2048
#define DTT    192   // DHC + DRR
#define NTOK   (BBB*SEQ)   // 4096

union FragA { v16bf v; v8bf h[2]; };

// Async 16B global->LDS copy (CDNA5 GLOBAL_LOAD_ASYNC_TO_LDS_B128, GV mode).
// vdst = LDS byte address (wave-relative; low 32 bits of generic shared ptr),
// vaddr = 64-bit global address. Tracked by ASYNCcnt.
__device__ __forceinline__ void async_stage_b128(const bf16_t* gsrc, bf16_t* lds_dst) {
  unsigned lds_off = (unsigned)(unsigned long long)lds_dst;
  asm volatile("global_load_async_to_lds_b128 %0, %1, off"
               :: "v"(lds_off), "v"(gsrc) : "memory");
}
__device__ __forceinline__ void wait_asynccnt0() {
  asm volatile("s_wait_asynccnt 0" ::: "memory");
}

// ---------------------------------------------------------------------------
// fp32 -> bf16 elementwise
__global__ void k_f32_to_bf16(const float* __restrict__ in, bf16_t* __restrict__ out, int n) {
  int i = blockIdx.x * blockDim.x + threadIdx.x;
  if (i < n) out[i] = (bf16_t)in[i];
}

// fp32 [rows,cols] -> bf16 [cols,rows] (weight transpose so B-fragment K is contiguous)
__global__ void k_transpose_to_bf16(const float* __restrict__ in, bf16_t* __restrict__ out,
                                    int rows, int cols) {
  int i = blockIdx.x * blockDim.x + threadIdx.x;
  if (i >= rows * cols) return;
  int r = i / cols, c = i % cols;
  out[(size_t)c * rows + r] = (bf16_t)in[i];
}

// ---------------------------------------------------------------------------
// bf16 GEMM: C[M,N] = A[M,K] * Bt[N,K]^T.
// Block = 128 threads (4 waves) computing a 64x32 C tile:
//   wave w -> rows m0+16w .. m0+16w+15, all waves share cols n0..n0+31.
// B panel (32 rows x 32 k = 2KB) staged in LDS with async copies, double
// buffered; A fragments loaded straight from global (b128 clauses).
// Fragments use contiguous k = 16*(lane/16) + s in both A and B (A 16x32 bf16, B 32x16 bf16).
__global__ void __launch_bounds__(128)
k_gemm_bf16(const bf16_t* __restrict__ A, const bf16_t* __restrict__ Bt,
            float* __restrict__ Cf, bf16_t* __restrict__ Cb,
            int M, int N, int K) {
  __shared__ __align__(16) bf16_t Bsh[2][32 * 32];   // [buf][row][k] 2KB each

  const int tid  = threadIdx.x;
  const int wave = tid >> 5;
  const int lane = tid & 31;
  const int r    = lane & 15;
  const int hi   = lane >> 4;

  const int n0 = blockIdx.x * 32;
  const int m0 = blockIdx.y * 64 + wave * 16;

  const bf16_t* Arow = A + (size_t)(m0 + r) * K;

  // staging assignment: thread copies 8 bf16 (16B): row n0 + tid/4, col (tid&3)*8
  const int srow = tid >> 2;
  const int scol = (tid & 3) * 8;
  const bf16_t* Bsrc   = Bt + (size_t)(n0 + srow) * K + scol;
  bf16_t*       Bdst0  = &Bsh[0][srow * 32 + scol];
  bf16_t*       Bdst1  = &Bsh[1][srow * 32 + scol];

  const int nchunks = K / 32;
  async_stage_b128(Bsrc, Bdst0);                    // prefetch chunk 0

  v8f acc0 = {}, acc1 = {};
  for (int kc = 0; kc < nchunks; ++kc) {
    wait_asynccnt0();
    __syncthreads();                                // panel[cur] visible to all waves
    const int cur = kc & 1;
    if (kc + 1 < nchunks)                           // prefetch next panel
      async_stage_b128(Bsrc + (size_t)(kc + 1) * 32, cur ? Bdst0 : Bdst1);

    const int k0 = kc * 32;
    FragA a;
    a.h[0] = *(const v8bf*)(Arow + k0 + hi * 16);
    a.h[1] = *(const v8bf*)(Arow + k0 + hi * 16 + 8);
    const bf16_t* Bl = Bsh[cur];
    v16bf b0 = *(const v16bf*)(Bl + r * 32 + hi * 16);
    v16bf b1 = *(const v16bf*)(Bl + (16 + r) * 32 + hi * 16);
    acc0 = __builtin_amdgcn_wmma_f32_16x16x32_bf16(false, a.v, false, b0,
                                                   (short)0, acc0, false, false);
    acc1 = __builtin_amdgcn_wmma_f32_16x16x32_bf16(false, a.v, false, b1,
                                                   (short)0, acc1, false, false);
    // reads are consumed by the WMMAs before the next top-of-loop barrier,
    // so one barrier per chunk is sufficient with double buffering.
  }

#pragma unroll
  for (int vv = 0; vv < 8; ++vv) {
    int row  = m0 + hi * 8 + vv;    // C layout: VGPR v -> M = v (+8 for lanes 16-31)
    int col0 = n0 + r;
    int col1 = n0 + 16 + r;
    if (Cf) {
      Cf[(size_t)row * N + col0] = acc0[vv];
      Cf[(size_t)row * N + col1] = acc1[vv];
    }
    if (Cb) {
      Cb[(size_t)row * N + col0] = (bf16_t)acc0[vv];
      Cb[(size_t)row * N + col1] = (bf16_t)acc1[vv];
    }
  }
}

// ---------------------------------------------------------------------------
// Pack Q = concat(Q_content, rope(Q_rope)) -> bf16 [B, NQ, S, DT]
__global__ void k_pack_q(const bf16_t* __restrict__ qc, const float* __restrict__ qr,
                         const float* __restrict__ cosc, const float* __restrict__ sinc,
                         const int* __restrict__ pos, bf16_t* __restrict__ Qp) {
  int i = blockIdx.x * blockDim.x + threadIdx.x;
  const int total = BBB * NQH * SEQ * DTT;
  if (i >= total) return;
  int d = i % DTT;
  int s = (i / DTT) % SEQ;
  int h = (i / (DTT * SEQ)) % NQH;
  int b = i / (DTT * SEQ * NQH);
  int tok = b * SEQ + s;
  if (d < DHC) {
    Qp[i] = qc[(size_t)tok * (NQH * DHC) + h * DHC + d];
  } else {
    int rr = d - DHC;
    const float* base = qr + (size_t)tok * (NQH * DRR) + h * DRR;
    float xv  = base[rr];
    float rot = (rr < DRR / 2) ? -base[rr + DRR / 2] : base[rr - DRR / 2];
    int p = pos[s];
    Qp[i] = (bf16_t)(xv * cosc[p * DRR + rr] + rot * sinc[p * DRR + rr]);
  }
}

// Pack K -> bf16 [B, NKV, S, DT]; also emit fp32 K_rope output [B,NKV,S,DR]
__global__ void k_pack_k(const bf16_t* __restrict__ kc, const float* __restrict__ kr,
                         const float* __restrict__ cosc, const float* __restrict__ sinc,
                         const int* __restrict__ pos, bf16_t* __restrict__ Kp,
                         float* __restrict__ krope_out) {
  int i = blockIdx.x * blockDim.x + threadIdx.x;
  const int total = BBB * NKVH * SEQ * DTT;
  if (i >= total) return;
  int d = i % DTT;
  int s = (i / DTT) % SEQ;
  int h = (i / (DTT * SEQ)) % NKVH;
  int b = i / (DTT * SEQ * NKVH);
  int tok = b * SEQ + s;
  if (d < DHC) {
    Kp[i] = kc[(size_t)tok * (NKVH * DHC) + h * DHC + d];
  } else {
    int rr = d - DHC;
    const float* base = kr + (size_t)tok * (NKVH * DRR) + h * DRR;
    float xv  = base[rr];
    float rot = (rr < DRR / 2) ? -base[rr + DRR / 2] : base[rr - DRR / 2];
    int p = pos[s];
    float val = xv * cosc[p * DRR + rr] + rot * sinc[p * DRR + rr];
    Kp[i] = (bf16_t)val;
    krope_out[(size_t)((b * NKVH + h) * SEQ + s) * DRR + rr] = val;
  }
}

// V [tok, NKV*DH] -> Vt bf16 [B, NKV, DH, S]  (so P*V B-operand K (=key) is contiguous)
__global__ void k_pack_vt(const bf16_t* __restrict__ v, bf16_t* __restrict__ Vt) {
  int i = blockIdx.x * blockDim.x + threadIdx.x;
  const int total = BBB * NKVH * DHC * SEQ;
  if (i >= total) return;
  int s  = i % SEQ;
  int dh = (i / SEQ) % DHC;
  int h  = (i / (SEQ * DHC)) % NKVH;
  int b  = i / (SEQ * DHC * NKVH);
  Vt[i] = v[(size_t)(b * SEQ + s) * (NKVH * DHC) + h * DHC + dh];
}

// ---------------------------------------------------------------------------
// Flash-style causal GQA attention. One wave per (head, 16-query tile).
// Per 32-key block: 12 score WMMAs (DT=192 -> 6 K-steps x 2 key subtiles),
// online softmax via shfl_xor row reductions, P staged through LDS
// (C-layout f32 -> A-layout bf16), 8 P*V WMMAs.
__global__ void __launch_bounds__(32)
k_mla_attn(const bf16_t* __restrict__ Q, const bf16_t* __restrict__ Kp,
           const bf16_t* __restrict__ Vt, bf16_t* __restrict__ attn) {
  __shared__ __align__(32) bf16_t Pl[16 * 32];

  const int q0   = blockIdx.x * 16;
  const int hq   = blockIdx.y;                 // 0 .. B*NQ-1
  const int b    = hq / NQH;
  const int h    = hq % NQH;
  const int kv   = h / (NQH / NKVH);
  const int lane = threadIdx.x;
  const int r    = lane & 15;
  const int hi   = lane >> 4;
  const float scale = 0.07216878364870323f;    // 1/sqrt(192)

  const bf16_t* Qh = Q  + (size_t)(b * NQH  + h)  * SEQ * DTT;
  const bf16_t* Kh = Kp + (size_t)(b * NKVH + kv) * SEQ * DTT;
  const bf16_t* Vh = Vt + (size_t)(b * NKVH + kv) * DHC * SEQ;

  // Preload Q fragments for this 16-row tile (6 chunks of K=32 over DT=192)
  FragA qf[6];
  const bf16_t* Qrow = Qh + (size_t)(q0 + r) * DTT;
#pragma unroll
  for (int c = 0; c < 6; ++c) {
    qf[c].h[0] = *(const v8bf*)(Qrow + c * 32 + hi * 16);
    qf[c].h[1] = *(const v8bf*)(Qrow + c * 32 + hi * 16 + 8);
  }

  v8f o[8] = {};                 // O accumulators, 8 dh-tiles of 16
  float rowm[8], rowl[8];
#pragma unroll
  for (int vv = 0; vv < 8; ++vv) { rowm[vv] = -1e30f; rowl[vv] = 0.0f; }

  const int kend = q0 + 15;      // causal: keys 0..q0+15
  for (int kt0 = 0; kt0 <= kend; kt0 += 32) {
    v8f s0 = {}, s1 = {};
    const bf16_t* K0row = Kh + (size_t)(kt0 + r) * DTT;
    const bf16_t* K1row = Kh + (size_t)(kt0 + 16 + r) * DTT;
#pragma unroll
    for (int c = 0; c < 6; ++c) {
      v16bf b0 = *(const v16bf*)(K0row + c * 32 + hi * 16);
      s0 = __builtin_amdgcn_wmma_f32_16x16x32_bf16(false, qf[c].v, false, b0,
                                                   (short)0, s0, false, false);
      v16bf b1 = *(const v16bf*)(K1row + c * 32 + hi * 16);
      s1 = __builtin_amdgcn_wmma_f32_16x16x32_bf16(false, qf[c].v, false, b1,
                                                   (short)0, s1, false, false);
    }

    // scale + causal mask + per-row block max
    float bm[8];
#pragma unroll
    for (int vv = 0; vv < 8; ++vv) {
      int row  = q0 + hi * 8 + vv;
      float a0 = s0[vv] * scale; if (kt0 + r      > row) a0 = -1e30f;
      float a1 = s1[vv] * scale; if (kt0 + 16 + r > row) a1 = -1e30f;
      s0[vv] = a0; s1[vv] = a1;
      bm[vv] = fmaxf(a0, a1);
    }
#pragma unroll
    for (int off = 1; off < 16; off <<= 1)
#pragma unroll
      for (int vv = 0; vv < 8; ++vv)
        bm[vv] = fmaxf(bm[vv], __shfl_xor(bm[vv], off, 32));

    // online softmax update, write P (bf16) to LDS in row-major [16][32]
    float bs[8];
#pragma unroll
    for (int vv = 0; vv < 8; ++vv) {
      float mnew  = fmaxf(rowm[vv], bm[vv]);
      float alpha = __expf(rowm[vv] - mnew);
      float p0 = __expf(s0[vv] - mnew);
      float p1 = __expf(s1[vv] - mnew);
      bs[vv]   = p0 + p1;
      rowm[vv] = mnew;
      rowl[vv] = rowl[vv] * alpha;
#pragma unroll
      for (int t = 0; t < 8; ++t) o[t][vv] *= alpha;
      Pl[(hi * 8 + vv) * 32 + r]      = (bf16_t)p0;
      Pl[(hi * 8 + vv) * 32 + 16 + r] = (bf16_t)p1;
    }
#pragma unroll
    for (int off = 1; off < 16; off <<= 1)
#pragma unroll
      for (int vv = 0; vv < 8; ++vv)
        bs[vv] += __shfl_xor(bs[vv], off, 32);
#pragma unroll
    for (int vv = 0; vv < 8; ++vv) rowl[vv] += bs[vv];

    // read P back as A fragment (single wave: LDS in-order, dscnt waits inserted)
    FragA pf;
    pf.h[0] = *(const v8bf*)(Pl + r * 32 + hi * 16);
    pf.h[1] = *(const v8bf*)(Pl + r * 32 + hi * 16 + 8);

    // O += P * V for each of the 8 dh tiles
#pragma unroll
    for (int t = 0; t < 8; ++t) {
      v16bf vb = *(const v16bf*)(Vh + (size_t)(t * 16 + r) * SEQ + kt0 + hi * 16);
      o[t] = __builtin_amdgcn_wmma_f32_16x16x32_bf16(false, pf.v, false, vb,
                                                     (short)0, o[t], false, false);
    }
  }

  // epilogue: O / l  -> attn bf16 [tok, NQ*DH]
#pragma unroll
  for (int t = 0; t < 8; ++t) {
#pragma unroll
    for (int vv = 0; vv < 8; ++vv) {
      int row = q0 + hi * 8 + vv;
      float val = o[t][vv] / rowl[vv];
      attn[(size_t)(b * SEQ + row) * (NQH * DHC) + h * DHC + t * 16 + r] = (bf16_t)val;
    }
  }
}

// ---------------------------------------------------------------------------
extern "C" void kernel_launch(void* const* d_in, const int* in_sizes, int n_in,
                              void* d_out, int out_size, void* d_ws, size_t ws_size,
                              hipStream_t stream) {
  (void)in_sizes; (void)n_in; (void)out_size; (void)ws_size;

  const float* x     = (const float*)d_in[0];
  const float* cosc  = (const float*)d_in[1];
  const float* sinc  = (const float*)d_in[2];
  const int*   pos   = (const int*)d_in[3];
  /* d_in[4] = attn_mask (causal tril) -- computed analytically in-kernel */
  const float* W_DKV = (const float*)d_in[5];
  const float* W_UK  = (const float*)d_in[6];
  const float* W_UV  = (const float*)d_in[7];
  const float* W_DQ  = (const float*)d_in[8];
  const float* W_UQ  = (const float*)d_in[9];
  const float* W_KR  = (const float*)d_in[10];
  const float* W_QR  = (const float*)d_in[11];
  const float* W_O   = (const float*)d_in[12];

  float* out_main  = (float*)d_out;                          // [B,S,D_MODEL]
  float* out_ckv   = out_main + (size_t)NTOK * DMODEL;       // [B,S,DKV]
  float* out_krope = out_ckv  + (size_t)NTOK * DKVL;         // [B,NKV,S,DR]

  // bump allocator on workspace (~154 MiB total)
  char* wsb = (char*)d_ws;
  size_t off = 0;
  auto alloc = [&](size_t bytes) -> void* {
    void* p = wsb + off;
    off = (off + bytes + 255) & ~(size_t)255;
    return p;
  };
  bf16_t* xb     = (bf16_t*)alloc((size_t)NTOK * DMODEL * 2);
  bf16_t* WtDKV  = (bf16_t*)alloc((size_t)DKVL * DMODEL * 2);
  bf16_t* WtUK   = (bf16_t*)alloc((size_t)(NKVH*DHC) * DKVL * 2);
  bf16_t* WtUV   = (bf16_t*)alloc((size_t)(NKVH*DHC) * DKVL * 2);
  bf16_t* WtDQ   = (bf16_t*)alloc((size_t)DQCL * DMODEL * 2);
  bf16_t* WtUQ   = (bf16_t*)alloc((size_t)(NQH*DHC) * DQCL * 2);
  bf16_t* WtKR   = (bf16_t*)alloc((size_t)(NKVH*DRR) * DMODEL * 2);
  bf16_t* WtQR   = (bf16_t*)alloc((size_t)(NQH*DRR) * DMODEL * 2);
  bf16_t* WtO    = (bf16_t*)alloc((size_t)DMODEL * (NQH*DHC) * 2);
  bf16_t* ckv_bf = (bf16_t*)alloc((size_t)NTOK * DKVL * 2);
  bf16_t* cq_bf  = (bf16_t*)alloc((size_t)NTOK * DQCL * 2);
  bf16_t* qc_bf  = (bf16_t*)alloc((size_t)NTOK * NQH * DHC * 2);
  float*  qr_f   = (float*) alloc((size_t)NTOK * NQH * DRR * 4);
  float*  kr_f   = (float*) alloc((size_t)NTOK * NKVH * DRR * 4);
  bf16_t* kc_bf  = (bf16_t*)alloc((size_t)NTOK * NKVH * DHC * 2);
  bf16_t* v_bf   = (bf16_t*)alloc((size_t)NTOK * NKVH * DHC * 2);
  bf16_t* Qp     = (bf16_t*)alloc((size_t)BBB * NQH  * SEQ * DTT * 2);
  bf16_t* Kpk    = (bf16_t*)alloc((size_t)BBB * NKVH * SEQ * DTT * 2);
  bf16_t* Vtw    = (bf16_t*)alloc((size_t)BBB * NKVH * DHC * SEQ * 2);
  bf16_t* attnb  = (bf16_t*)alloc((size_t)NTOK * NQH * DHC * 2);

  auto cdiv = [](int a, int b) { return (a + b - 1) / b; };
  const int T = 256;

  // ---- precision conversion (fp32 -> bf16; weights transposed to [N,K]) ----
  k_f32_to_bf16<<<cdiv(NTOK*DMODEL, T), T, 0, stream>>>(x, xb, NTOK*DMODEL);
  k_transpose_to_bf16<<<cdiv(DMODEL*DKVL, T),        T, 0, stream>>>(W_DKV, WtDKV, DMODEL, DKVL);
  k_transpose_to_bf16<<<cdiv(DKVL*(NKVH*DHC), T),    T, 0, stream>>>(W_UK,  WtUK,  DKVL, NKVH*DHC);
  k_transpose_to_bf16<<<cdiv(DKVL*(NKVH*DHC), T),    T, 0, stream>>>(W_UV,  WtUV,  DKVL, NKVH*DHC);
  k_transpose_to_bf16<<<cdiv(DMODEL*DQCL, T),        T, 0, stream>>>(W_DQ,  WtDQ,  DMODEL, DQCL);
  k_transpose_to_bf16<<<cdiv(DQCL*(NQH*DHC), T),     T, 0, stream>>>(W_UQ,  WtUQ,  DQCL, NQH*DHC);
  k_transpose_to_bf16<<<cdiv(DMODEL*(NKVH*DRR), T),  T, 0, stream>>>(W_KR,  WtKR,  DMODEL, NKVH*DRR);
  k_transpose_to_bf16<<<cdiv(DMODEL*(NQH*DRR), T),   T, 0, stream>>>(W_QR,  WtQR,  DMODEL, NQH*DRR);
  k_transpose_to_bf16<<<cdiv(DMODEL*DMODEL, T),      T, 0, stream>>>(W_O,   WtO,   NQH*DHC, DMODEL);

  // ---- projection GEMMs (WMMA, async-LDS staged B panels) ----
  // c_kv = x @ W_DKV  (fp32 -> output slot, bf16 -> downstream)
  k_gemm_bf16<<<dim3(DKVL/32, NTOK/64), 128, 0, stream>>>(xb, WtDKV, out_ckv, ckv_bf, NTOK, DKVL, DMODEL);
  // k_rope_lin = x @ W_KR
  k_gemm_bf16<<<dim3((NKVH*DRR)/32, NTOK/64), 128, 0, stream>>>(xb, WtKR, kr_f, nullptr, NTOK, NKVH*DRR, DMODEL);
  // c_q = x @ W_DQ
  k_gemm_bf16<<<dim3(DQCL/32, NTOK/64), 128, 0, stream>>>(xb, WtDQ, nullptr, cq_bf, NTOK, DQCL, DMODEL);
  // Q_content = c_q @ W_UQ
  k_gemm_bf16<<<dim3((NQH*DHC)/32, NTOK/64), 128, 0, stream>>>(cq_bf, WtUQ, nullptr, qc_bf, NTOK, NQH*DHC, DQCL);
  // Q_rope_lin = x @ W_QR
  k_gemm_bf16<<<dim3((NQH*DRR)/32, NTOK/64), 128, 0, stream>>>(xb, WtQR, qr_f, nullptr, NTOK, NQH*DRR, DMODEL);
  // K_content = c_kv @ W_UK ; V = c_kv @ W_UV
  k_gemm_bf16<<<dim3((NKVH*DHC)/32, NTOK/64), 128, 0, stream>>>(ckv_bf, WtUK, nullptr, kc_bf, NTOK, NKVH*DHC, DKVL);
  k_gemm_bf16<<<dim3((NKVH*DHC)/32, NTOK/64), 128, 0, stream>>>(ckv_bf, WtUV, nullptr, v_bf, NTOK, NKVH*DHC, DKVL);

  // ---- RoPE + layout packing ----
  k_pack_q<<<cdiv(BBB*NQH*SEQ*DTT, T), T, 0, stream>>>(qc_bf, qr_f, cosc, sinc, pos, Qp);
  k_pack_k<<<cdiv(BBB*NKVH*SEQ*DTT, T), T, 0, stream>>>(kc_bf, kr_f, cosc, sinc, pos, Kpk, out_krope);
  k_pack_vt<<<cdiv(BBB*NKVH*DHC*SEQ, T), T, 0, stream>>>(v_bf, Vtw);

  // ---- causal GQA attention (WMMA flash) ----
  k_mla_attn<<<dim3(SEQ/16, BBB*NQH), 32, 0, stream>>>(Qp, Kpk, Vtw, attnb);

  // ---- output projection ----
  k_gemm_bf16<<<dim3(DMODEL/32, NTOK/64), 128, 0, stream>>>(attnb, WtO, out_main, nullptr, NTOK, DMODEL, NQH*DHC);
}
